// gscalar_sym_33990371181236
// MI455X (gfx1250) — matrix operand order repaired and checked by reading, not compiled, not run on hardware
//
#include <hip/hip_runtime.h>
#include <hip/hip_bf16.h>
#include <math.h>

// ---------------------------------------------------------------------------
// gscalar symmetry-average:  out[b] = mean_{8 x 256 variants} G(variant(x[b]))
// G(z) = tanh(z @ W1 + b1) . w2 + b2
// Fused 131072x1024x256 GEMM on v_wmma_f32_16x16x32_f16; A-rows (signed
// permutations of x under roll/flip/parity) synthesized in registers from LDS.
// ---------------------------------------------------------------------------

typedef __attribute__((ext_vector_type(16))) _Float16 v16h;
typedef __attribute__((ext_vector_type(8)))  float    v8f;

#define LL   16          // lattice side
#define K    256         // L*L, GEMM K
#define H    1024        // hidden, GEMM N
#define NB   64          // batch
#define NTILES (H / 16)  // 64 N-tiles
#define KCH  (K / 32)    // 8 K-chunks of 32
#define NVAR 2048        // 8 * 256 variants per batch

// Branchless tanh: native V_TANH_F32 if the builtin exists, else
// tanh(z) = 1 - 2/(exp2(2*log2e*z)+1)  (v_exp_f32 + v_rcp_f32, no EXEC
// divergence; limits: z->+inf => rcp(inf)=0 => +1, z->-inf => 1-2 = -1).
__device__ __forceinline__ float fast_tanh(float z) {
#if __has_builtin(__builtin_amdgcn_tanhf)
    return __builtin_amdgcn_tanhf(z);
#else
    float e;
#if __has_builtin(__builtin_amdgcn_exp2f)
    e = __builtin_amdgcn_exp2f(z * 2.885390082f);   // 2*log2(e)
#else
    e = __expf(2.0f * z);
#endif
#if __has_builtin(__builtin_amdgcn_rcpf)
    float r = __builtin_amdgcn_rcpf(e + 1.0f);
#else
    float r = __frcp_rn(e + 1.0f);
#endif
    return 1.0f - 2.0f * r;
#endif
}

// ---------------------------------------------------------------------------
// Prep: repack W1 (f32, [K=256][H=1024] row-major) into f16 fragments laid out
// in per-lane WMMA B-operand order: contiguous 32B per lane, 1KB per wave per
// (nt,kc) block.  p = ((nt*8 + kc)*32 + lane)*16 + t2, where lane half
// (lane>>4) selects K 0..15 vs 16..31 of the 32-K chunk; VGPR t = t2>>1 holds
// packed K = kc*32 + half*16 + 2t (+ t2&1), N = nt*16 + lane&15.
// ---------------------------------------------------------------------------
__global__ __launch_bounds__(256) void gsym_prep_w1(const float* __restrict__ W1,
                                                    _Float16* __restrict__ W1p) {
    int p    = blockIdx.x * 256 + threadIdx.x;      // 0 .. 262143
    int t2   = p & 15;
    int lane = (p >> 4) & 31;
    int kc   = (p >> 9) & 7;
    int nt   = p >> 12;
    int k = kc * 32 + (lane >> 4) * 16 + (t2 & ~1) + (t2 & 1);
    int n = nt * 16 + (lane & 15);
    W1p[p] = (_Float16)W1[k * H + n];
}

// ---------------------------------------------------------------------------
// Main: grid = 1024 workgroups of 256 threads (8 waves).
//   wg -> (s = wg>>2, bg = wg&3): shift s = (sx,sy), batch rows bg*16..+15
//   wave w -> symmetry variant v (sign, flip_i, flip_j)
// Each wave: build A (16 rows x 256 K, f16) once; 64 N-tiles x 8 WMMA each.
// ---------------------------------------------------------------------------
__global__ __launch_bounds__(256) void gsym_main(const float* __restrict__ x,
                                                 const _Float16* __restrict__ W1p,
                                                 const float* __restrict__ b1,
                                                 const float* __restrict__ w2,
                                                 float* __restrict__ partials) {
    __shared__ float lx[16 * K];     // 16 batch rows of x, 16 KB
    __shared__ float lpart[8 * 16];  // per-wave row partial sums

    const int tid  = threadIdx.x;
    const int wave = tid >> 5;
    const int lane = tid & 31;
    const int wg   = blockIdx.x;
    const int s    = wg >> 2;
    const int bg   = wg & 3;
    const int sx   = s >> 4;
    const int sy   = s & 15;

    // Stage the 16 x-rows (4096 floats) for this batch group, coalesced.
    const float* xsrc = x + (size_t)(bg * 16) * K;
    for (int i = tid; i < 16 * K; i += 256) lx[i] = xsrc[i];
    __syncthreads();

    // Variant: allv = [id, flip_i, flip_j, flip_ij, then negated four]
    const int   v   = wave;
    const float sgn = (v < 4) ? 1.0f : -1.0f;
    const int   fi  = (v & 1) ? 15 : 0;   // flip along Lx (axis 1)
    const int   fj  = (v & 2) ? 15 : 0;   // flip along Ly (axis 2)
    const int   m    = lane & 15;         // A-matrix row for this lane
    const int   half = lane >> 4;

    // Build A fragments in contiguous K order (matching packed B):
    // lane<16 : VGPR t -> K = 32kc + 2t..;
    // lane>=16: K = 32kc + 16 + 2t..
    v16h a[KCH];
    for (int kc = 0; kc < KCH; ++kc) {
        for (int t = 0; t < 8; ++t) {
            int k0 = kc * 32 + half * 16 + t * 2;
#pragma unroll
            for (int u = 0; u < 2; ++u) {
                int k  = k0 + u;
                int i  = (k >> 4) ^ fi;        // flip (15-i == i^15)
                int j  = (k & 15) ^ fj;
                int si = (i - sx) & 15;        // periodic roll
                int sj = (j - sy) & 15;
                a[kc][2 * t + u] = (_Float16)(sgn * lx[m * K + si * LL + sj]);
            }
        }
    }

    const int ncol = lane & 15;
    float racc[8];
#pragma unroll
    for (int q = 0; q < 8; ++q) racc[q] = 0.0f;

    for (int nt = 0; nt < NTILES; ++nt) {
        const _Float16* bp = W1p + ((size_t)nt * KCH * 32 + lane) * 16;
        // prefetch next tile's fragment stream (global_prefetch_b8)
        if (nt + 1 < NTILES)
            __builtin_prefetch(bp + (size_t)KCH * 32 * 16, 0, 1);

        v8f acc = {};
#pragma unroll
        for (int kc = 0; kc < KCH; ++kc) {
            v16h bfrag = *(const v16h*)(bp + (size_t)kc * 32 * 16);
            acc = __builtin_amdgcn_wmma_f32_16x16x32_f16(
                /*neg_a=*/false, a[kc], /*neg_b=*/false, bfrag,
                /*c_mod=*/(short)0, acc, /*reuse_a=*/false, /*reuse_b=*/false);
        }

        const float b1n = b1[nt * 16 + ncol];
        const float w2n = w2[nt * 16 + ncol];
#pragma unroll
        for (int q = 0; q < 8; ++q)
            racc[q] += fast_tanh(acc[q] + b1n) * w2n;  // branchless epilogue
    }

    // C/D layout: VGPR q, lane -> row M = q + 8*half, col = lane&15.
    // Sum each row over its 16 lanes (deterministic shfl-xor tree).
#pragma unroll
    for (int q = 0; q < 8; ++q) {
        float sv = racc[q];
        sv += __shfl_xor(sv, 1, 32);
        sv += __shfl_xor(sv, 2, 32);
        sv += __shfl_xor(sv, 4, 32);
        sv += __shfl_xor(sv, 8, 32);
        racc[q] = sv;
    }
    if ((lane & 15) == 0) {
        int rbase = half * 8;
        for (int q = 0; q < 8; ++q) lpart[wave * 16 + rbase + q] = racc[q];
    }
    __syncthreads();

    // Fixed-order cross-wave (variant) sum -> per-WG partial; deterministic.
    if (tid < 16) {
        float accv = 0.0f;
        for (int w = 0; w < 8; ++w) accv += lpart[w * 16 + tid];
        partials[(size_t)wg * 16 + tid] = accv;
    }
}

// ---------------------------------------------------------------------------
// Final reduce: out[b] = (1/2048) * sum_s partials[(s*4 + b/16)*16 + b%16] + b2
// ---------------------------------------------------------------------------
__global__ void gsym_final(const float* __restrict__ partials,
                           const float* __restrict__ b2,
                           float* __restrict__ out) {
    int b = threadIdx.x;              // 64 threads
    int bg = b >> 4, row = b & 15;
    float acc = 0.0f;
    for (int s = 0; s < 256; ++s)
        acc += partials[(size_t)(s * 4 + bg) * 16 + row];
    out[b] = acc * (1.0f / (float)NVAR) + b2[0];
}

// ---------------------------------------------------------------------------
extern "C" void kernel_launch(void* const* d_in, const int* in_sizes, int n_in,
                              void* d_out, int out_size, void* d_ws, size_t ws_size,
                              hipStream_t stream) {
    const float* x  = (const float*)d_in[0];   // [64,16,16]
    const float* W1 = (const float*)d_in[1];   // [256,1024]
    const float* b1 = (const float*)d_in[2];   // [1024]
    const float* w2 = (const float*)d_in[3];   // [1024]
    const float* b2 = (const float*)d_in[4];   // [1]
    float* out = (float*)d_out;                // [64]

    _Float16* W1p     = (_Float16*)d_ws;                       // 512 KB packed W1
    float*    partial = (float*)((char*)d_ws + (size_t)K * H * sizeof(_Float16));
                                                               // 1024*16 f32 = 64 KB

    // 1) repack W1 -> f16 WMMA-B fragment order
    gsym_prep_w1<<<dim3((K * H) / 256), dim3(256), 0, stream>>>(W1, W1p);
    // 2) fused symmetry-GEMM + tanh + dot(w2): 1024 WGs x 8 waves
    gsym_main<<<dim3(1024), dim3(256), 0, stream>>>(x, W1p, b1, w2, partial);
    // 3) deterministic mean + bias
    gsym_final<<<dim3(1), dim3(64), 0, stream>>>(partial, b2, out);
}
